// BERT_54709293416848
// MI455X (gfx1250) — matrix operand order repaired and checked by reading, not compiled, not run on hardware
//
#include <hip/hip_runtime.h>
#include <hip/hip_bf16.h>
#include <math.h>
#include <stdint.h>

// ---------------------------------------------------------------------------
// BERT-base forward on gfx1250 (MI455X).
// Compute-bound (~2.9 TFLOP) -> bf16 WMMA w/ f32 accumulate.
// Weights pre-transposed+converted once to bf16 [N][K]; GEMM tiles (A and B,
// both 128x32 bf16) are DMA'd into double-buffered LDS by the Tensor Data
// Mover (tensor_load_to_lds, TENSORcnt) with D# padding reproducing the
// 80B-row LDS stride, overlapping TDM fill with v_wmma_f32_16x16x32_bf16.
// ---------------------------------------------------------------------------

typedef unsigned short u16;
typedef __attribute__((ext_vector_type(16))) __bf16 v16bf;
typedef __attribute__((ext_vector_type(8)))  float  v8f;
typedef __attribute__((ext_vector_type(4)))  unsigned su4;
typedef __attribute__((ext_vector_type(8)))  unsigned su8;

__device__ __forceinline__ u16 f2bf(float f) {
    unsigned u = __float_as_uint(f);
    u += 0x7fffu + ((u >> 16) & 1u);      // round-to-nearest-even
    return (u16)(u >> 16);
}

union FragB16 { v16bf v; uint4 q[2]; u16 u[16]; };

// ---------------------------------------------------------------------------
// TDM: load a 128-row x 32-elem bf16 tile (row stride = strideK elems) into
// LDS with 16B padding after each 64B row (-> 80B LDS row stride).
// D# group0: count=1, lds_addr, global_addr, type=2.
// D# group1: data_size=2B, pad_enable, pad_interval=16dw, pad_amount=4dw,
//            tensor_dim0/1 (OOB), tile_dim0=32, tile_dim1=128, dim0_stride=K.
// ---------------------------------------------------------------------------
__device__ __forceinline__ void tdm_load_128x32(unsigned lds_addr, const u16* g,
                                                unsigned rem_k, unsigned rem_rows,
                                                unsigned strideK)
{
    unsigned long long ga = (unsigned long long)(uintptr_t)g;
    su4 g0;
    g0[0] = 1u;                                     // count=1, user mode
    g0[1] = lds_addr;                               // LDS byte address
    g0[2] = (unsigned)ga;                           // global_addr[31:0]
    g0[3] = (unsigned)((ga >> 32) & 0x01ffffffu) | (2u << 30);  // [56:32] | type=2
    su8 g1;
    g1[0] = 0x06D10000u;  // data_size=1(2B)<<16 | pad_en<<20 | interval=3<<22 | amount=3<<25
    g1[1] = (rem_k & 0xffffu) << 16;                          // tensor_dim0[15:0]
    g1[2] = (rem_k >> 16) | ((rem_rows & 0xffffu) << 16);     // td0[31:16] | td1[15:0]
    g1[3] = (rem_rows >> 16) | (32u << 16);                   // td1[31:16] | tile_dim0=32
    g1[4] = 128u;                                             // tile_dim1=128, tile_dim2=0
    g1[5] = strideK;                                          // tensor_dim0_stride[31:0]
    g1[6] = 0u;
    g1[7] = 0u;
    asm volatile("tensor_load_to_lds %0, %1" :: "s"(g0), "s"(g1) : "memory");
}

// ---------------------------------------------------------------------------
// Weight pre-pass: W[L][K][N] f32 -> Wt[L][N][K] bf16 (once per launch).
// ---------------------------------------------------------------------------
__global__ __launch_bounds__(256)
void transpose_w_kernel(const float* __restrict__ W, u16* __restrict__ Wt,
                        int K, int N)
{
    __shared__ u16 tile[32][33];
    const size_t mat = (size_t)K * N;
    const float* Wl = W + (size_t)blockIdx.z * mat;
    u16* Wtl = Wt + (size_t)blockIdx.z * mat;
    const int n0 = blockIdx.x * 32, k0 = blockIdx.y * 32;
    const int tx = threadIdx.x & 31, ty = threadIdx.x >> 5;
    #pragma unroll
    for (int yy = ty; yy < 32; yy += 8)
        tile[yy][tx] = f2bf(Wl[(size_t)(k0 + yy) * N + n0 + tx]);
    __syncthreads();
    #pragma unroll
    for (int yy = ty; yy < 32; yy += 8)
        Wtl[(size_t)(n0 + yy) * K + k0 + tx] = tile[tx][yy];
}

// ---------------------------------------------------------------------------
// GEMM: C[M,N] = act(A_bf16[M,K] @ Wt_bf16[N,K]^T + bias[N])
// 128x128 block tile, BK=32, 8 waves (2x4), wave tile 64x32.
// TDM double-buffered LDS fill overlapped with WMMA.
// ---------------------------------------------------------------------------
__global__ __launch_bounds__(256)
void gemm_bf16_kernel(const u16* __restrict__ A, const u16* __restrict__ Wt,
                      const float* __restrict__ bias,
                      float* __restrict__ Cf, u16* __restrict__ Cb,
                      int M, int K, int N, int gelu)
{
    __shared__ __align__(16) u16 As[2][128][40];   // [m][k], 80B row (16B multiple)
    __shared__ __align__(16) u16 Bs[2][128][40];   // [n][k]

    const int tid  = threadIdx.x;
    const int lane = tid & 31;
    const int wid  = tid >> 5;
    const int half = lane >> 4;      // 0/1
    const int l15  = lane & 15;
    const int waveM = wid >> 2;      // 0..1 -> 64-row patch
    const int waveN = wid & 3;       // 0..3 -> 32-col patch
    const int m0 = blockIdx.y * 128;
    const int n0 = blockIdx.x * 128;

    const unsigned bufBytes = 128 * 40 * 2;
    const unsigned asB = (unsigned)(uintptr_t)&As[0][0][0];
    const unsigned bsB = (unsigned)(uintptr_t)&Bs[0][0][0];
    const bool tdmWave = (__builtin_amdgcn_readfirstlane(wid) == 0);

    if (tdmWave) {   // prime buffer 0
        tdm_load_128x32(asB, A  + (size_t)m0 * K, K, M - m0, K);
        tdm_load_128x32(bsB, Wt + (size_t)n0 * K, K, N - n0, K);
        __builtin_amdgcn_s_wait_tensorcnt(0);
    }
    __syncthreads();

    v8f acc[4][2];
    #pragma unroll
    for (int i = 0; i < 4; ++i)
        #pragma unroll
        for (int j = 0; j < 2; ++j) acc[i][j] = (v8f)(0.0f);

    int cur = 0;
    for (int k0 = 0; k0 < K; k0 += 32) {
        const int kn = k0 + 32;
        if (tdmWave && kn < K) {  // async prefetch next K-slice into other buffer
            tdm_load_128x32(asB + (cur ^ 1) * bufBytes,
                            A  + (size_t)m0 * K + kn, K - kn, M - m0, K);
            tdm_load_128x32(bsB + (cur ^ 1) * bufBytes,
                            Wt + (size_t)n0 * K + kn, K - kn, N - n0, K);
        }

        FragB16 afrag[4], bfrag[2];
        #pragma unroll
        for (int i = 0; i < 4; ++i) {  // A frag per ISA 16-bit A layout
            int m = waveM * 64 + i * 16 + l15;
            afrag[i].q[0] = *(const uint4*)&As[cur][m][half * 8];
            afrag[i].q[1] = *(const uint4*)&As[cur][m][16 + half * 8];
        }
        #pragma unroll
        for (int j = 0; j < 2; ++j) {  // B frag: same K layout as A frag
            int n = waveN * 32 + j * 16 + l15;
            bfrag[j].q[0] = *(const uint4*)&Bs[cur][n][half * 8];
            bfrag[j].q[1] = *(const uint4*)&Bs[cur][n][16 + half * 8];
        }
        #pragma unroll
        for (int i = 0; i < 4; ++i)
            #pragma unroll
            for (int j = 0; j < 2; ++j)
                acc[i][j] = __builtin_amdgcn_wmma_f32_16x16x32_bf16(
                    false, afrag[i].v, false, bfrag[j].v, (short)0, acc[i][j],
                    false, false);

        if (tdmWave) __builtin_amdgcn_s_wait_tensorcnt(0);
        __syncthreads();
        cur ^= 1;
    }

    #pragma unroll
    for (int i = 0; i < 4; ++i) {
        #pragma unroll
        for (int j = 0; j < 2; ++j) {
            #pragma unroll
            for (int r = 0; r < 8; ++r) {
                int m = m0 + waveM * 64 + i * 16 + half * 8 + r;
                int n = n0 + waveN * 32 + j * 16 + l15;
                float v = acc[i][j][r] + bias[n];
                if (gelu) v = 0.5f * v * (1.0f + erff(v * 0.70710678118654752f));
                if (Cf) Cf[(size_t)m * N + n] = v;
                if (Cb) Cb[(size_t)m * N + n] = f2bf(v);
            }
        }
    }
}

// ---------------------------------------------------------------------------
// Attention: one block per (b,h). K and V^T resident in LDS; 4 query chunks
// of 32 rows. scores = QK^T/8 + mask -> softmax -> P@V. WMMA throughout.
// ---------------------------------------------------------------------------
__global__ __launch_bounds__(256)
void attn_kernel(const u16* __restrict__ Q, const u16* __restrict__ Kg,
                 const u16* __restrict__ Vg, const int* __restrict__ xids,
                 u16* __restrict__ ctx)
{
    __shared__ __align__(16) u16   Ks[128][72];   // [s_k][dk]
    __shared__ __align__(16) u16   Vt[64][136];   // [dk][s_k]
    __shared__ __align__(16) u16   Qs[32][72];    // [s_q][dk]
    __shared__ __align__(16) float Sc[32][132];   // scores / probs
    __shared__ float maskv[128];

    const int tid  = threadIdx.x;
    const int lane = tid & 31;
    const int wid  = tid >> 5;
    const int half = lane >> 4;
    const int l15  = lane & 15;
    const int bh = blockIdx.x;
    const int b  = bh / 12;
    const int h  = bh % 12;
    const int D  = 768;

    if (tid < 128) maskv[tid] = (xids[b * 128 + tid] > 0) ? 0.0f : -1e9f;

    { // stage K rows and V transposed
        int s    = tid >> 1;
        int part = (tid & 1) * 32;   // halfs
        const uint4* ksrc = (const uint4*)(Kg + ((size_t)(b * 128 + s) * D + h * 64 + part));
        uint4 k0 = ksrc[0], k1 = ksrc[1], k2 = ksrc[2], k3 = ksrc[3];
        *(uint4*)&Ks[s][part]      = k0;
        *(uint4*)&Ks[s][part + 8]  = k1;
        *(uint4*)&Ks[s][part + 16] = k2;
        *(uint4*)&Ks[s][part + 24] = k3;

        uint4 vbuf[4];
        const uint4* vsrc = (const uint4*)(Vg + ((size_t)(b * 128 + s) * D + h * 64 + part));
        vbuf[0] = vsrc[0]; vbuf[1] = vsrc[1]; vbuf[2] = vsrc[2]; vbuf[3] = vsrc[3];
        const u16* vu = (const u16*)vbuf;
        #pragma unroll
        for (int ii = 0; ii < 32; ++ii)
            Vt[part + ii][s] = vu[ii];
    }
    __syncthreads();

    for (int qc = 0; qc < 4; ++qc) {
        { // stage 32 query rows
            int s    = tid >> 3;
            int part = (tid & 7) * 8;
            const uint4* qsrc =
                (const uint4*)(Q + ((size_t)(b * 128 + qc * 32 + s) * D + h * 64 + part));
            *(uint4*)&Qs[s][part] = qsrc[0];
        }
        __syncthreads();

        // scores: 2x8 grid of 16x16 tiles, 2 per wave; K-dim = 64 (2 WMMA steps)
        #pragma unroll
        for (int t = 0; t < 2; ++t) {
            int idx = wid + t * 8;
            int tm = idx >> 3, tn = idx & 7;
            v8f sacc = (v8f)(0.0f);
            #pragma unroll
            for (int ks = 0; ks < 2; ++ks) {
                FragB16 a, bf;
                int m = tm * 16 + l15;
                a.q[0] = *(const uint4*)&Qs[m][ks * 32 + half * 8];
                a.q[1] = *(const uint4*)&Qs[m][ks * 32 + 16 + half * 8];
                int n = tn * 16 + l15;
                bf.q[0] = *(const uint4*)&Ks[n][ks * 32 + half * 8];
                bf.q[1] = *(const uint4*)&Ks[n][ks * 32 + 16 + half * 8];
                sacc = __builtin_amdgcn_wmma_f32_16x16x32_bf16(
                    false, a.v, false, bf.v, (short)0, sacc, false, false);
            }
            #pragma unroll
            for (int r = 0; r < 8; ++r) {
                int m = tm * 16 + half * 8 + r;
                int n = tn * 16 + l15;
                Sc[m][n] = sacc[r] * 0.125f + maskv[n];
            }
        }
        __syncthreads();

        if (tid < 32) { // softmax, one row per thread
            float mx = -3.0e38f;
            for (int j = 0; j < 128; ++j) mx = fmaxf(mx, Sc[tid][j]);
            float sum = 0.0f;
            for (int j = 0; j < 128; ++j) {
                float e = __expf(Sc[tid][j] - mx);
                Sc[tid][j] = e; sum += e;
            }
            float inv = 1.0f / sum;
            for (int j = 0; j < 128; ++j) Sc[tid][j] *= inv;
        }
        __syncthreads();

        { // ctx = P @ V : 2x4 tiles, 1 per wave; K-dim = 128 (4 WMMA steps)
            int tm = wid >> 2, tn = wid & 3;
            v8f cacc = (v8f)(0.0f);
            #pragma unroll
            for (int ks = 0; ks < 4; ++ks) {
                FragB16 a, bf;
                int m = tm * 16 + l15;
                #pragma unroll
                for (int e = 0; e < 16; ++e) { // A-frag element e -> K per ISA layout
                    int k = ks * 32 + (e >> 3) * 16 + half * 8 + (e & 7);
                    a.u[e] = f2bf(Sc[m][k]);
                }
                int n = tn * 16 + l15;
                bf.q[0] = *(const uint4*)&Vt[n][ks * 32 + half * 8];
                bf.q[1] = *(const uint4*)&Vt[n][ks * 32 + 16 + half * 8];
                cacc = __builtin_amdgcn_wmma_f32_16x16x32_bf16(
                    false, a.v, false, bf.v, (short)0, cacc, false, false);
            }
            #pragma unroll
            for (int r = 0; r < 8; ++r) {
                int m  = qc * 32 + tm * 16 + half * 8 + r;
                int dk = tn * 16 + l15;
                ctx[((size_t)(b * 128 + m) * D) + h * 64 + dk] = f2bf(cacc[r]);
            }
        }
        __syncthreads();
    }
}

// ---------------------------------------------------------------------------
// Fused residual-add + LayerNorm; writes fp32 stream + bf16 shadow.
// ---------------------------------------------------------------------------
__global__ __launch_bounds__(256)
void add_ln_kernel(const float* __restrict__ resid, const float* __restrict__ add,
                   const float* __restrict__ g, const float* __restrict__ be,
                   float* __restrict__ outf, u16* __restrict__ outb)
{
    __shared__ float red[256];
    const int row = blockIdx.x;
    const int tid = threadIdx.x;
    float xv[3];
    #pragma unroll
    for (int q = 0; q < 3; ++q) {
        int c = tid + q * 256;
        size_t idx = (size_t)row * 768 + c;
        xv[q] = resid[idx] + add[idx];
    }
    red[tid] = xv[0] + xv[1] + xv[2];
    __syncthreads();
    for (int off = 128; off > 0; off >>= 1) {
        if (tid < off) red[tid] += red[tid + off];
        __syncthreads();
    }
    float mu = red[0] * (1.0f / 768.0f);
    __syncthreads();
    float v = 0.0f;
    #pragma unroll
    for (int q = 0; q < 3; ++q) { float d = xv[q] - mu; v += d * d; }
    red[tid] = v;
    __syncthreads();
    for (int off = 128; off > 0; off >>= 1) {
        if (tid < off) red[tid] += red[tid + off];
        __syncthreads();
    }
    float rstd = rsqrtf(red[0] * (1.0f / 768.0f) + 1e-5f);
    #pragma unroll
    for (int q = 0; q < 3; ++q) {
        int c = tid + q * 256;
        size_t idx = (size_t)row * 768 + c;
        float y = (xv[q] - mu) * rstd * g[c] + be[c];
        outf[idx] = y;
        outb[idx] = f2bf(y);
    }
}

// ---------------------------------------------------------------------------
// Embedding: tok + seg + PE (PE recomputed in-kernel, matching the reference's
// 10000**(2*i/D) exponent with i = 0,2,4,...).
// ---------------------------------------------------------------------------
__global__ __launch_bounds__(256)
void embed_kernel(const int* __restrict__ x, const int* __restrict__ seg,
                  const float* __restrict__ tokE, const float* __restrict__ segE,
                  float* __restrict__ hf, u16* __restrict__ hb)
{
    const int row = blockIdx.x;       // b*S + s
    const int s   = row & 127;
    const int tid = threadIdx.x;
    const int t   = x[row];
    const int gidx = seg[row];
    for (int c = tid; c < 768; c += 256) {
        int   k2  = c & ~1;
        float freq = __powf(10000.0f, -((float)(2 * k2)) / 768.0f);
        float arg  = (float)s * freq;
        float pe   = (c & 1) ? __cosf(arg) : __sinf(arg);
        float v = tokE[(size_t)t * 768 + c] + segE[(size_t)gidx * 768 + c] + pe;
        size_t idx = (size_t)row * 768 + c;
        hf[idx] = v;
        hb[idx] = f2bf(v);
    }
}

// ---------------------------------------------------------------------------
extern "C" void kernel_launch(void* const* d_in, const int* in_sizes, int n_in,
                              void* d_out, int out_size, void* d_ws, size_t ws_size,
                              hipStream_t stream)
{
    const int*   x    = (const int*)  d_in[0];
    const int*   seg  = (const int*)  d_in[1];
    const float* tokE = (const float*)d_in[2];
    const float* segE = (const float*)d_in[3];
    const float* Wq = (const float*)d_in[4];
    const float* bq = (const float*)d_in[5];
    const float* Wk = (const float*)d_in[6];
    const float* bk = (const float*)d_in[7];
    const float* Wv = (const float*)d_in[8];
    const float* bv = (const float*)d_in[9];
    const float* Wo = (const float*)d_in[10];
    const float* bo = (const float*)d_in[11];
    const float* lng = (const float*)d_in[12];
    const float* lnb = (const float*)d_in[13];
    const float* W1 = (const float*)d_in[14];
    const float* b1 = (const float*)d_in[15];
    const float* W2 = (const float*)d_in[16];
    const float* b2 = (const float*)d_in[17];
    (void)in_sizes; (void)n_in; (void)ws_size;

    const int Mrows = 128 * 128;  // B*S = 16384
    char*  wsb = (char*)d_ws;
    size_t off = 0;
    auto alloc = [&](size_t bytes) -> void* {
        void* p = wsb + off;
        off += (bytes + 255) & ~(size_t)255;
        return p;
    };
    float* h_f   = (float*)alloc((size_t)Mrows * 768 * 4);
    u16*   h_b   = (u16*)  alloc((size_t)Mrows * 768 * 2);
    float* h1_f  = (float*)alloc((size_t)Mrows * 768 * 4);
    u16*   h1_b  = (u16*)  alloc((size_t)Mrows * 768 * 2);
    float* tmp_f = (float*)alloc((size_t)Mrows * 768 * 4);  // attn-out / ffn-out
    u16*   q_b   = (u16*)  alloc((size_t)Mrows * 768 * 2);
    u16*   k_b   = (u16*)  alloc((size_t)Mrows * 768 * 2);
    u16*   v_b   = (u16*)  alloc((size_t)Mrows * 768 * 2);
    u16*   c_b   = (u16*)  alloc((size_t)Mrows * 768 * 2);
    u16*   mid_b = (u16*)  alloc((size_t)Mrows * 3072 * 2);
    // transposed bf16 weights [L][N][K]
    u16*   wq_t  = (u16*)  alloc((size_t)12 * 768 * 768 * 2);
    u16*   wk_t  = (u16*)  alloc((size_t)12 * 768 * 768 * 2);
    u16*   wv_t  = (u16*)  alloc((size_t)12 * 768 * 768 * 2);
    u16*   wo_t  = (u16*)  alloc((size_t)12 * 768 * 768 * 2);
    u16*   w1_t  = (u16*)  alloc((size_t)12 * 3072 * 768 * 2);
    u16*   w2_t  = (u16*)  alloc((size_t)12 * 768 * 3072 * 2);

    dim3 blk(256);
    dim3 g768(768 / 128, Mrows / 128);
    dim3 g3072(3072 / 128, Mrows / 128);

    // one-time weight transpose/convert (L2-friendly, ~20us of HBM traffic)
    transpose_w_kernel<<<dim3(768/32, 768/32, 12),  blk, 0, stream>>>(Wq, wq_t, 768, 768);
    transpose_w_kernel<<<dim3(768/32, 768/32, 12),  blk, 0, stream>>>(Wk, wk_t, 768, 768);
    transpose_w_kernel<<<dim3(768/32, 768/32, 12),  blk, 0, stream>>>(Wv, wv_t, 768, 768);
    transpose_w_kernel<<<dim3(768/32, 768/32, 12),  blk, 0, stream>>>(Wo, wo_t, 768, 768);
    transpose_w_kernel<<<dim3(3072/32, 768/32, 12), blk, 0, stream>>>(W1, w1_t, 768, 3072);
    transpose_w_kernel<<<dim3(768/32, 3072/32, 12), blk, 0, stream>>>(W2, w2_t, 3072, 768);

    embed_kernel<<<Mrows, blk, 0, stream>>>(x, seg, tokE, segE, h_f, h_b);

    for (int l = 0; l < 12; ++l) {
        const u16* wql = wq_t + (size_t)l * 768 * 768;
        const u16* wkl = wk_t + (size_t)l * 768 * 768;
        const u16* wvl = wv_t + (size_t)l * 768 * 768;
        const u16* wol = wo_t + (size_t)l * 768 * 768;
        const u16* w1l = w1_t + (size_t)l * 3072 * 768;
        const u16* w2l = w2_t + (size_t)l * 768 * 3072;
        const float* bql = bq + (size_t)l * 768;
        const float* bkl = bk + (size_t)l * 768;
        const float* bvl = bv + (size_t)l * 768;
        const float* bol = bo + (size_t)l * 768;
        const float* b1l = b1 + (size_t)l * 3072;
        const float* b2l = b2 + (size_t)l * 768;
        const float* gl  = lng + (size_t)l * 768;
        const float* bel = lnb + (size_t)l * 768;

        gemm_bf16_kernel<<<g768, blk, 0, stream>>>(h_b, wql, bql, nullptr, q_b,
                                                   Mrows, 768, 768, 0);
        gemm_bf16_kernel<<<g768, blk, 0, stream>>>(h_b, wkl, bkl, nullptr, k_b,
                                                   Mrows, 768, 768, 0);
        gemm_bf16_kernel<<<g768, blk, 0, stream>>>(h_b, wvl, bvl, nullptr, v_b,
                                                   Mrows, 768, 768, 0);
        attn_kernel<<<128 * 12, blk, 0, stream>>>(q_b, k_b, v_b, x, c_b);
        gemm_bf16_kernel<<<g768, blk, 0, stream>>>(c_b, wol, bol, tmp_f, nullptr,
                                                   Mrows, 768, 768, 0);
        add_ln_kernel<<<Mrows, blk, 0, stream>>>(h_f, tmp_f, gl, bel, h1_f, h1_b);
        gemm_bf16_kernel<<<g3072, blk, 0, stream>>>(h1_b, w1l, b1l, nullptr, mid_b,
                                                    Mrows, 768, 3072, 1);
        gemm_bf16_kernel<<<g768, blk, 0, stream>>>(mid_b, w2l, b2l, tmp_f, nullptr,
                                                   Mrows, 3072, 768, 0);
        add_ln_kernel<<<Mrows, blk, 0, stream>>>(h1_f, tmp_f, gl, bel, h_f, h_b);
    }

    hipMemcpyAsync(d_out, h_f, (size_t)out_size * sizeof(float),
                   hipMemcpyDeviceToDevice, stream);
}
